// GRNN_84550726189135
// MI455X (gfx1250) — compile-verified, matrix operand order checked by reading, not run on hardware
//
#include <hip/hip_runtime.h>
#include <hip/hip_bf16.h>

typedef __attribute__((ext_vector_type(16))) _Float16 v16h;
typedef __attribute__((ext_vector_type(8)))  float    v8f;

#define D 256            // feature dim (both layers)
#define LDS_PAD 8        // f16 pad -> row stride 528B (16B-aligned, banks rotate by 4)

// ---------------- degree / normalization ----------------
__global__ void k_deg_init(float* deg, int n) {
    int i = blockIdx.x * blockDim.x + threadIdx.x;
    if (i < n) deg[i] = 1.0f;                    // +1 self-loop
}

__global__ void k_deg_count(const long long* __restrict__ dst, float* deg, int e) {
    int i = blockIdx.x * blockDim.x + threadIdx.x;
    if (i < e) atomicAdd(&deg[(int)dst[i]], 1.0f);
}

__global__ void k_isq(float* deg, int n) {       // in-place: deg -> deg^{-1/2}
    int i = blockIdx.x * blockDim.x + threadIdx.x;
    if (i < n) deg[i] = rsqrtf(deg[i]);
}

// ---------------- edge prep: int32 indices + per-edge norm (shared by both layers) ----
__global__ void k_edge_prep(const long long* __restrict__ src64,
                            const long long* __restrict__ dst64,
                            const float* __restrict__ isq,
                            int* __restrict__ src32, int* __restrict__ dst32,
                            float* __restrict__ ew, int e) {
    int i = blockIdx.x * blockDim.x + threadIdx.x;
    if (i < e) {
        int s = (int)src64[i], d = (int)dst64[i];
        src32[i] = s;
        dst32[i] = d;
        ew[i]    = isq[s] * isq[d];
    }
}

// ---------------- fp32 -> fp16 convert (x only) ----------------
__global__ void k_cvt_f16(const float* __restrict__ src, _Float16* __restrict__ dstp, int n) {
    int i = blockIdx.x * blockDim.x + threadIdx.x;
    if (i < n) dstp[i] = (_Float16)src[i];
}

// ---------------- W repack: fp32 [256x256] -> WMMA B-fragment layout (f16) -----------
// Bp index: ((tile*8 + kt)*32 + lane)*16 + j ; j=2i   -> W[k(i)  ][col]
//                                              j=2i+1 -> W[k(i)+1][col]
// col = tile*16 + (lane&15); half = lane>>4; k(i) = kt*32 + (i<4 ? half*8+2i : 16+half*8+2(i-4))
__global__ void k_repack_w(const float* __restrict__ W, _Float16* __restrict__ Bp) {
    int t = blockIdx.x * blockDim.x + threadIdx.x;   // 0..4095 = (tile, kt, lane)
    if (t >= 16 * 8 * 32) return;
    int lane = t & 31;
    int kt   = (t >> 5) & 7;
    int tile = t >> 8;
    int col  = tile * 16 + (lane & 15);
    int half = lane >> 4;
    _Float16* out = Bp + (size_t)t * 16;
    #pragma unroll
    for (int i = 0; i < 8; ++i) {
        int k = kt * 32 + ((i < 4) ? (half * 8 + 2 * i) : (16 + half * 8 + 2 * (i - 4)));
        out[2 * i]     = (_Float16)W[(size_t)k       * D + col];
        out[2 * i + 1] = (_Float16)W[(size_t)(k + 1) * D + col];
    }
}

// ---------------- WMMA GEMM + fused self-loop ----------------
// C[N x 256] = A[N x 256] * B ; agg = C * isq^2. One block per 16-row strip (N % 16 == 0).
// 8 waves; wave w owns column tiles 2w, 2w+1. K-loop: 8 steps of K=32, fully unrolled.
__global__ __launch_bounds__(256) void k_gemm(const _Float16* __restrict__ A,
                                              const _Float16* __restrict__ Bp,
                                              const float* __restrict__ isq,
                                              float* __restrict__ C,
                                              float* __restrict__ agg) {
    __shared__ _Float16 As[16][D + LDS_PAD];
    const int rb = blockIdx.x * 16;
    const int t  = threadIdx.x;

    // cooperative stage of the 16x256 f16 A-strip (two b128 loads / thread)
    {
        int row = t >> 4;             // 0..15
        int kb  = (t & 15) << 4;      // 0,16,...,240
        const uint4* s4 = (const uint4*)(A + (size_t)(rb + row) * D + kb);
        uint4 p0 = s4[0];
        uint4 p1 = s4[1];
        *(uint4*)&As[row][kb]     = p0;
        *(uint4*)&As[row][kb + 8] = p1;
    }
    __syncthreads();

    const int wave = t >> 5;          // 0..7
    const int lane = t & 31;
    const int lrow = lane & 15;       // A row / B-C column within tile
    const int half = lane >> 4;       // K-half (A/B), M-half (C/D)

    v8f c0 = {}; v8f c1 = {};
    const int col0 = wave * 32 + lrow;
    const int col1 = col0 + 16;
    const uint4* bp0 = (const uint4*)(Bp + (((size_t)(2 * wave)     * 8) * 32 + lane) * 16);
    const uint4* bp1 = (const uint4*)(Bp + (((size_t)(2 * wave + 1) * 8) * 32 + lane) * 16);

    #pragma unroll
    for (int kt = 0; kt < 8; ++kt) {  // K = 8 * 32 = 256
        union { v16h v; uint4 q[2]; } a, b0, b1;
        // A fragment: two contiguous 16B runs in LDS per ISA 7.12.2 layout
        const _Float16* ar = &As[lrow][kt * 32 + half * 8];
        a.q[0] = *(const uint4*)ar;         // K = kt*32 + half*8 + 0..7
        a.q[1] = *(const uint4*)(ar + 16);  // K = kt*32 + half*8 + 16..23
        // B fragments: fragment-ready, 32B contiguous per lane (L2-resident)
        b0.q[0] = bp0[kt * 64 + 0];         // (kt*32 + lane) uint4-pairs
        b0.q[1] = bp0[kt * 64 + 1];
        b1.q[0] = bp1[kt * 64 + 0];
        b1.q[1] = bp1[kt * 64 + 1];
        c0 = __builtin_amdgcn_wmma_f32_16x16x32_f16(false, a.v, false, b0.v,
                                                    (short)0, c0, false, false);
        c1 = __builtin_amdgcn_wmma_f32_16x16x32_f16(false, a.v, false, b1.v,
                                                    (short)0, c1, false, false);
    }

    #pragma unroll
    for (int i = 0; i < 8; ++i) {     // C/D: VGPR i -> row i + 8*half, col = lane&15
        int row = rb + i + half * 8;
        float sv = isq[row];
        float inv = sv * sv;          // deg^{-1} (self-loop norm)
        size_t o0 = (size_t)row * D + col0;
        size_t o1 = (size_t)row * D + col1;
        C[o0]   = c0[i];
        C[o1]   = c1[i];
        agg[o0] = c0[i] * inv;
        agg[o1] = c1[i] * inv;
    }
}

// ---------------- edge scatter: agg[dst] += ew * h[src] ----------------
// 4 edges per 256-thread block; 64 lanes per edge, float4 gather, 4 f32 atomics.
__global__ __launch_bounds__(256) void k_scatter(const int* __restrict__ src,
                                                 const int* __restrict__ dst,
                                                 const float* __restrict__ ew,
                                                 const float* __restrict__ h,
                                                 float* __restrict__ agg, int e) {
    int eid = blockIdx.x * 4 + (threadIdx.x >> 6);
    if (eid >= e) return;
    int q = (threadIdx.x & 63) << 2;                 // feature offset 0..252
    int s = src[eid];
    int d = dst[eid];
    float w = ew[eid];
    const float4 hv = *(const float4*)(h + (size_t)s * D + q);
    float* out = agg + (size_t)d * D + q;
    atomicAdd(out + 0, hv.x * w);
    atomicAdd(out + 1, hv.y * w);
    atomicAdd(out + 2, hv.z * w);
    atomicAdd(out + 3, hv.w * w);
}

// ---------------- bias + ReLU ----------------
__global__ void k_finish_f16(const float* __restrict__ agg, const float* __restrict__ bias,
                             _Float16* __restrict__ out, int n) {  // feeds next layer
    int i = blockIdx.x * blockDim.x + threadIdx.x;
    if (i < n * D) {
        float v = agg[i] + bias[i & (D - 1)];
        out[i] = (_Float16)(v > 0.0f ? v : 0.0f);
    }
}

__global__ void k_finish_f32(const float* __restrict__ agg, const float* __restrict__ bias,
                             float* __restrict__ out, int n) {     // final output
    int i = blockIdx.x * blockDim.x + threadIdx.x;
    if (i < n * D) {
        float v = agg[i] + bias[i & (D - 1)];
        out[i] = v > 0.0f ? v : 0.0f;
    }
}

extern "C" void kernel_launch(void* const* d_in, const int* in_sizes, int n_in,
                              void* d_out, int out_size, void* d_ws, size_t ws_size,
                              hipStream_t stream) {
    const float*     x  = (const float*)d_in[0];
    const long long* ei = (const long long*)d_in[1];   // int64 [2, E]
    const float*     W1 = (const float*)d_in[2];
    const float*     b1 = (const float*)d_in[3];
    const float*     W2 = (const float*)d_in[4];
    const float*     b2 = (const float*)d_in[5];
    float* out = (float*)d_out;

    const int N = in_sizes[0] / D;          // 50000 (divisible by 16)
    const int E = in_sizes[1] / 2;          // 800000
    const long long* src64 = ei;
    const long long* dst64 = ei + E;

    // workspace layout
    char* wsp = (char*)d_ws;
    float*     isq   = (float*)wsp;                                 // N f32
    float*     h     = (float*)(wsp + (size_t)256 * 1024);          // N*D f32
    float*     agg   = h + (size_t)N * D;                           // N*D f32
    _Float16*  xin   = (_Float16*)(agg + (size_t)N * D);            // N*D f16
    _Float16*  wfp   = xin + (size_t)N * D;                         // D*D f16 (frag layout)
    int*       src32 = (int*)(wfp + (size_t)D * D);                 // E i32
    int*       dst32 = src32 + E;                                   // E i32
    float*     ew    = (float*)(dst32 + E);                         // E f32

    const int ND = N * D;
    dim3 blk(256);
    dim3 gN((N + 255) / 256), gE((E + 255) / 256), gND((ND + 255) / 256);
    dim3 gGemm(N / 16);
    dim3 gScat((E + 3) / 4);
    dim3 gRep(16);                         // 4096 threads for W repack

    // normalization + edge metadata (once, shared by both layers)
    k_deg_init  <<<gN, blk, 0, stream>>>(isq, N);
    k_deg_count <<<gE, blk, 0, stream>>>(dst64, isq, E);
    k_isq       <<<gN, blk, 0, stream>>>(isq, N);
    k_edge_prep <<<gE, blk, 0, stream>>>(src64, dst64, isq, src32, dst32, ew, E);

    // ---- layer 1 ----
    k_cvt_f16   <<<gND, blk, 0, stream>>>(x, xin, ND);
    k_repack_w  <<<gRep, blk, 0, stream>>>(W1, wfp);
    k_gemm      <<<gGemm, blk, 0, stream>>>(xin, wfp, isq, h, agg);
    k_scatter   <<<gScat, blk, 0, stream>>>(src32, dst32, ew, h, agg, E);
    k_finish_f16<<<gND, blk, 0, stream>>>(agg, b1, xin, N);   // relu'd f16 input of L2

    // ---- layer 2 ----
    k_repack_w  <<<gRep, blk, 0, stream>>>(W2, wfp);
    k_gemm      <<<gGemm, blk, 0, stream>>>(xin, wfp, isq, h, agg);
    k_scatter   <<<gScat, blk, 0, stream>>>(src32, dst32, ew, h, agg, E);
    k_finish_f32<<<gND, blk, 0, stream>>>(agg, b2, out, N);
}
